// OpPolicyNet_45749991637041
// MI455X (gfx1250) — matrix operand order repaired and checked by reading, not compiled, not run on hardware
//
#include <hip/hip_runtime.h>
#include <math.h>

typedef __attribute__((ext_vector_type(16))) _Float16 v16h;
typedef __attribute__((ext_vector_type(8)))  _Float16 v8h;
typedef __attribute__((ext_vector_type(8)))  float    v8f;
typedef __attribute__((ext_vector_type(2)))  float    v2f;

#define HW 256
#define BATCH 32

// ---------------------------------------------------------------------------
// Kernel 1: per-(batch,color) count / ysum / xsum  -> stats [B,40]
// ---------------------------------------------------------------------------
__global__ __launch_bounds__(256)
void stats_kernel(const int* __restrict__ grid, float* __restrict__ stats) {
    const int b = blockIdx.x, tid = threadIdx.x;
    int cnt[10], sy[10], sx[10];
#pragma unroll
    for (int c = 0; c < 10; ++c) { cnt[c] = 0; sy[c] = 0; sx[c] = 0; }
    for (int i = tid; i < HW * HW; i += 256) {
        int y = i >> 8, x = i & 255;
        int g = grid[b * HW * HW + i];
#pragma unroll
        for (int c = 0; c < 10; ++c) {
            int m = (g == c) ? 1 : 0;
            cnt[c] += m; sy[c] += m * y; sx[c] += m * x;
        }
    }
    __shared__ int tot[30];
    if (tid < 30) tot[tid] = 0;
    __syncthreads();
#pragma unroll
    for (int c = 0; c < 10; ++c) {
        atomicAdd(&tot[c],      cnt[c]);
        atomicAdd(&tot[10 + c], sy[c]);
        atomicAdd(&tot[20 + c], sx[c]);
    }
    __syncthreads();
    if (tid < 10) {
        float count = (float)tot[tid];
        float ym = (count > 0.f) ? (float)tot[10 + tid] / count : 0.f;
        float xm = (count > 0.f) ? (float)tot[20 + tid] / count : 0.f;
        float* o = stats + b * 40 + tid * 4;
        o[0] = count; o[1] = count; o[2] = ym; o[3] = xm;
    }
}

// ---------------------------------------------------------------------------
// Kernel 2: fused conv1(gather) -> conv2(WMMA f16) -> ReLU -> 32x32 avg pool
// One block = one 32x32 output tile of one image.  grid = (8,8,32), 512 thr.
// ---------------------------------------------------------------------------
__global__ __launch_bounds__(512)
void conv_fused_kernel(const int* __restrict__ grid,
                       const float* __restrict__ w1, const float* __restrict__ b1,
                       const float* __restrict__ w2, const float* __restrict__ b2,
                       float* __restrict__ pooled) {
    __shared__ int      gT[36 * 36];          // grid tile, halo 2, -1 = OOB
    __shared__ float    w1l[10 * 9 * 32];     // [color][tap][outch]
    __shared__ float    b1l[32];
    __shared__ _Float16 act[34 * 34 * 32];    // conv1 activation halo, ch-last
    __shared__ _Float16 bfr[9 * 4 * 32 * 16]; // conv2 B fragments [tap][nsub][lane][16]
    __shared__ float    poolL[16 * 64];

    const int tid = threadIdx.x;
    const int tx = blockIdx.x, ty = blockIdx.y, b = blockIdx.z;
    const int y0 = ty * 32, x0 = tx * 32;

    // stage 0: grid tile with halo 2
    for (int i = tid; i < 36 * 36; i += 512) {
        int gy = i / 36, gx = i % 36;
        int y = y0 - 2 + gy, x = x0 - 2 + gx;
        gT[i] = (y >= 0 && y < HW && x >= 0 && x < HW) ? grid[(b * HW + y) * HW + x] : -1;
    }
    // conv1 weights -> [g][tap][c]   (src layout [c][g][ky][kx])
    for (int i = tid; i < 2880; i += 512) {
        int c = i & 31, rest = i >> 5, tap = rest % 9, g = rest / 9;
        w1l[i] = w1[(c * 10 + g) * 9 + tap];
    }
    if (tid < 32) b1l[tid] = b1[tid];
    // conv2 weights -> WMMA B fragments.  For wmma_f32_16x16x32_f16 B (32x16):
    // lane L holds n = nsub*16 + (L&15); half j holds k=8*(L>>4)+j (j<8), 16+8*(L>>4)+j-8 (j>=8).
    for (int i = tid; i < 9 * 4 * 32 * 16; i += 512) {
        int j = i & 15, lane = (i >> 4) & 31, nsub = (i >> 9) & 3, tap = i >> 11;
        int k = (j < 8) ? 8 * (lane >> 4) + j : 16 + 8 * (lane >> 4) + (j - 8);
        int n = nsub * 16 + (lane & 15);
        bfr[i] = (_Float16)w2[(n * 32 + k) * 9 + tap];   // [n][k][ky][kx]
    }
    __syncthreads();

    // stage 1: conv1 gather -> act halo (34x34), zero outside image (SAME pad)
    for (int p = tid; p < 34 * 34; p += 512) {
        int py = p / 34, px = p % 34;
        int y = y0 - 1 + py, x = x0 - 1 + px;
        if (y >= 0 && y < HW && x >= 0 && x < HW) {
            float acc[32];
#pragma unroll
            for (int c = 0; c < 32; ++c) acc[c] = b1l[c];
#pragma unroll
            for (int ky = 0; ky < 3; ++ky)
#pragma unroll
                for (int kx = 0; kx < 3; ++kx) {
                    int g = gT[(py + ky) * 36 + (px + kx)];
                    if (g >= 0) {
                        const float* wr = &w1l[(g * 9 + ky * 3 + kx) * 32];
#pragma unroll
                        for (int c = 0; c < 32; ++c) acc[c] += wr[c];
                    }
                }
#pragma unroll
            for (int c = 0; c < 32; ++c) act[p * 32 + c] = (_Float16)fmaxf(acc[c], 0.f);
        } else {
#pragma unroll
            for (int c = 0; c < 32; ++c) act[p * 32 + c] = (_Float16)0.f;
        }
    }
    __syncthreads();

    // stage 2: implicit GEMM, M=1024 pixels, N=64, K=288 (9 taps * 32ch)
    const int wave = tid >> 5, lane = tid & 31;
    const int hi = lane >> 4, lm = lane & 15;
    for (int nsub = 0; nsub < 4; ++nsub) {
        float bias = b2[nsub * 16 + lm];
        v8f accs[4];
#pragma unroll
        for (int ms = 0; ms < 4; ++ms)
#pragma unroll
            for (int r = 0; r < 8; ++r) accs[ms][r] = bias;

        for (int tap = 0; tap < 9; ++tap) {
            v16h bf = *(const v16h*)&bfr[((tap * 4 + nsub) * 32 + lane) * 16];
            int ky = tap / 3, kx = tap % 3;
#pragma unroll
            for (int ms = 0; ms < 4; ++ms) {
                int m = wave * 64 + ms * 16 + lm;
                int pyo = m >> 5, pxo = m & 31;
                // A frag: lane holds row m; halves 0..7 -> k=hi*8+j, 8..15 -> k=16+hi*8+j
                const v8h* pix = (const v8h*)&act[((pyo + ky) * 34 + (pxo + kx)) * 32];
                v8h lo = pix[hi];
                v8h hh = pix[2 + hi];
                v16h af = __builtin_shufflevector(lo, hh, 0, 1, 2, 3, 4, 5, 6, 7,
                                                  8, 9, 10, 11, 12, 13, 14, 15);
                accs[ms] = __builtin_amdgcn_wmma_f32_16x16x32_f16(
                    false, af, false, bf, (short)0, accs[ms], false, false);
            }
        }
        // ReLU + pool partial: lane L covers n=lm, rows r+8*hi of each msub tile
        float s = 0.f;
#pragma unroll
        for (int ms = 0; ms < 4; ++ms)
#pragma unroll
            for (int r = 0; r < 8; ++r) s += fmaxf(accs[ms][r], 0.f);
        s += __shfl_xor(s, 16);
        if (lm == lane) poolL[wave * 64 + nsub * 16 + lm] = s;  // lanes 0..15
    }
    __syncthreads();
    if (tid < 64) {
        float s = 0.f;
        for (int w = 0; w < 16; ++w) s += poolL[w * 64 + tid];
        pooled[b * 4096 + tid * 64 + ty * 8 + tx] = s * (1.0f / 1024.0f);
    }
}

// ---------------------------------------------------------------------------
// Kernel 3: tiny feature MLPs (slot / prog / rel / size / theme) -> ctx cols
// One block per batch row, 64 threads.
// ---------------------------------------------------------------------------
struct SFParams {
    const float *stats, *relin, *sizein, *themein;
    const float *s0W, *s0b, *s1W, *s1b, *s2W, *s2b;
    const float *p0b, *p1W, *p1b, *p2W, *p2b;
    const float *r0W, *r0b, *r1W, *r1b;
    const float *z0W, *z0b, *z1W, *z1b;
    const float *t0W, *t0b, *t1W, *t1b;
    float* ctx;
};

__global__ __launch_bounds__(64)
void small_feats_kernel(SFParams P) {
    const int b = blockIdx.x, j = threadIdx.x;
    __shared__ float sA[64], sB[64];
    float* ctx = P.ctx + b * 320;

    // ---- slots: 40 -> 64 -> 32 -> 32  (relu after first two) ----
    {
        const float* in = P.stats + b * 40;
        float a = P.s0b[j];
        for (int i = 0; i < 40; ++i) a += in[i] * P.s0W[i * 64 + j];
        sA[j] = fmaxf(a, 0.f);
        __syncthreads();
        if (j < 32) {
            float a2 = P.s1b[j];
            for (int i = 0; i < 64; ++i) a2 += sA[i] * P.s1W[i * 32 + j];
            sB[j] = fmaxf(a2, 0.f);
        }
        __syncthreads();
        if (j < 32) {
            float a3 = P.s2b[j];
            for (int i = 0; i < 32; ++i) a3 += sB[i] * P.s2W[i * 32 + j];
            ctx[128 + j] = a3;
        }
        __syncthreads();
    }
    // ---- prog: input is zeros -> layer0 out = b0 ----
    {
        sA[j] = fmaxf(P.p0b[j], 0.f);
        __syncthreads();
        if (j < 32) {
            float a2 = P.p1b[j];
            for (int i = 0; i < 64; ++i) a2 += sA[i] * P.p1W[i * 32 + j];
            sB[j] = fmaxf(a2, 0.f);
        }
        __syncthreads();
        if (j < 32) {
            float a3 = P.p2b[j];
            for (int i = 0; i < 32; ++i) a3 += sB[i] * P.p2W[i * 32 + j];
            ctx[160 + j] = a3;
        }
        __syncthreads();
    }
    // ---- rel: 64 -> 64 -> 64 ----
    {
        const float* in = P.relin + b * 64;
        float a = P.r0b[j];
        for (int i = 0; i < 64; ++i) a += in[i] * P.r0W[i * 64 + j];
        sA[j] = fmaxf(a, 0.f);
        __syncthreads();
        float a2 = P.r1b[j];
        for (int i = 0; i < 64; ++i) a2 += sA[i] * P.r1W[i * 64 + j];
        ctx[192 + j] = a2;
        __syncthreads();
    }
    // ---- size: 4 -> 16 -> 32 ----
    {
        const float* in = P.sizein + b * 4;
        if (j < 16) {
            float a = P.z0b[j];
            for (int i = 0; i < 4; ++i) a += in[i] * P.z0W[i * 16 + j];
            sA[j] = fmaxf(a, 0.f);
        }
        __syncthreads();
        if (j < 32) {
            float a2 = P.z1b[j];
            for (int i = 0; i < 16; ++i) a2 += sA[i] * P.z1W[i * 32 + j];
            ctx[256 + j] = a2;
        }
        __syncthreads();
    }
    // ---- theme: 10 -> 32 -> 32 ----
    {
        const float* in = P.themein + b * 10;
        if (j < 32) {
            float a = P.t0b[j];
            for (int i = 0; i < 10; ++i) a += in[i] * P.t0W[i * 32 + j];
            sA[j] = fmaxf(a, 0.f);
        }
        __syncthreads();
        if (j < 32) {
            float a2 = P.t1b[j];
            for (int i = 0; i < 32; ++i) a2 += sA[i] * P.t1W[i * 32 + j];
            ctx[288 + j] = a2;
        }
    }
}

// ---------------------------------------------------------------------------
// Kernel 4: generic GEMM  C[M=32,N] = A[32,K] @ W[K,N] (+bias +bias2) (+act)
// via exact-f32 WMMA 16x16x4.  One wave per 16x16 C tile.
// ---------------------------------------------------------------------------
__global__ __launch_bounds__(32)
void gemm_wmma_kernel(const float* __restrict__ A, int lda,
                      const float* __restrict__ W,
                      const float* __restrict__ bias, const float* __restrict__ bias2,
                      float* __restrict__ C, int ldc, int N, int K, int act) {
    const int n0 = blockIdx.x * 16, m0 = blockIdx.y * 16;
    const int lane = threadIdx.x & 31;
    const int lm = lane & 15, hi = lane >> 4;
    const int n = n0 + lm;
    const bool nok = (n < N);
    const float* arow = A + (m0 + lm) * lda;

    v8f acc = {};
    for (int kk = 0; kk < K; kk += 4) {
        int ka = kk + 2 * hi;                 // A/B: VGPR v -> k = v + 2*hi
        v2f a; a[0] = arow[ka]; a[1] = arow[ka + 1];
        v2f bv;
        bv[0] = nok ? W[ka * N + n] : 0.f;
        bv[1] = nok ? W[(ka + 1) * N + n] : 0.f;
        acc = __builtin_amdgcn_wmma_f32_16x16x4_f32(
            false, a, false, bv, (short)0, acc, false, false);
    }
    float badd = 0.f;
    if (bias)  badd += bias[nok ? n : 0];
    if (bias2) badd += bias2[nok ? n : 0];
    if (nok) {
#pragma unroll
        for (int r = 0; r < 8; ++r) {
            int m = m0 + r + 8 * hi;          // C/D: vgpr r, lane-half hi -> row
            float v = acc[r] + badd;
            if (act == 1) v = fmaxf(v, 0.f);
            else if (act == 2) v = 1.f / (1.f + expf(-v));
            C[m * ldc + n] = v;
        }
    }
}

// ---------------------------------------------------------------------------
// Kernel 5: h = LayerNorm(h + r) * g + b   (rows of 512, 32 blocks x 256 thr)
// ---------------------------------------------------------------------------
__global__ __launch_bounds__(256)
void add_ln_kernel(float* __restrict__ h, const float* __restrict__ r,
                   const float* __restrict__ g, const float* __restrict__ bb) {
    const int b = blockIdx.x, tid = threadIdx.x;
    float x0 = h[b * 512 + tid]       + r[b * 512 + tid];
    float x1 = h[b * 512 + 256 + tid] + r[b * 512 + 256 + tid];
    __shared__ float red[256], red2[256];
    red[tid] = x0 + x1;
    red2[tid] = x0 * x0 + x1 * x1;
    __syncthreads();
    for (int s = 128; s > 0; s >>= 1) {
        if (tid < s) { red[tid] += red[tid + s]; red2[tid] += red2[tid + s]; }
        __syncthreads();
    }
    float mean = red[0] * (1.f / 512.f);
    float var = red2[0] * (1.f / 512.f) - mean * mean;
    float inv = 1.f / sqrtf(var + 1e-5f);
    h[b * 512 + tid]       = (x0 - mean) * inv * g[tid]       + bb[tid];
    h[b * 512 + 256 + tid] = (x1 - mean) * inv * g[256 + tid] + bb[256 + tid];
}

// ---------------------------------------------------------------------------
// Host side
// ---------------------------------------------------------------------------
extern "C" void kernel_launch(void* const* d_in, const int* in_sizes, int n_in,
                              void* d_out, int out_size, void* d_ws, size_t ws_size,
                              hipStream_t stream) {
    (void)in_sizes; (void)n_in; (void)out_size; (void)ws_size;
    // pytree-flattened leaves (dict keys sorted, lists in order, "W" < "b"):
    auto Pf = [&](int i) { return (const float*)d_in[i]; };
    const int* grid = (const int*)d_in[0];
    const float* rel_in   = Pf(1);
    const float* size_in  = Pf(2);
    const float* theme_in = Pf(3);
    // params: 4..136 (see analysis for the index map)
    float* out = (float*)d_out;

    float* ws = (float*)d_ws;
    float* pooled = ws;               // 32*4096
    float* stats  = ws + 131072;      // 32*40
    float* ctx    = ws + 132352;      // 32*320
    float* h      = ws + 142592;      // 32*512
    float* t0     = ws + 158976;      // 32*2048
    float* t1     = ws + 224512;      // 32*512
    float* t2     = ws + 240896;      // 32*512

    auto GEMM = [&](const float* A, int lda, const float* W, const float* bias,
                    const float* bias2, float* C, int ldc, int N, int K, int act) {
        dim3 g((N + 15) / 16, 2);
        gemm_wmma_kernel<<<g, 32, 0, stream>>>(A, lda, W, bias, bias2, C, ldc, N, K, act);
    };

    // 1) slot stats
    stats_kernel<<<BATCH, 256, 0, stream>>>(grid, stats);
    // 2) fused conv1+conv2+relu+pool  (conv1_b=12, conv1_w=13, conv2_b=14, conv2_w=15)
    conv_fused_kernel<<<dim3(8, 8, BATCH), 512, 0, stream>>>(
        grid, Pf(13), Pf(12), Pf(15), Pf(14), pooled);
    // 3) tiny MLP features -> ctx[:,128:320]
    SFParams sp;
    sp.stats = stats; sp.relin = rel_in; sp.sizein = size_in; sp.themein = theme_in;
    sp.s0W = Pf(119); sp.s0b = Pf(120); sp.s1W = Pf(121); sp.s1b = Pf(122);
    sp.s2W = Pf(123); sp.s2b = Pf(124);
    sp.p0b = Pf(102); sp.p1W = Pf(103); sp.p1b = Pf(104); sp.p2W = Pf(105); sp.p2b = Pf(106);
    sp.r0W = Pf(107); sp.r0b = Pf(108); sp.r1W = Pf(109); sp.r1b = Pf(110);
    sp.z0W = Pf(115); sp.z0b = Pf(116); sp.z1W = Pf(117); sp.z1b = Pf(118);
    sp.t0W = Pf(129); sp.t0b = Pf(130); sp.t1W = Pf(131); sp.t1b = Pf(132);
    sp.ctx = ctx;
    small_feats_kernel<<<BATCH, 64, 0, stream>>>(sp);
    // 4) grid_fc (W=20,b=21): pooled[32,4096] -> ctx[:,0:128]
    GEMM(pooled, 4096, Pf(20), Pf(21), nullptr, ctx, 320, 128, 4096, 0);
    // 5) in_proj (W=22,b=23) + pos[0] (100): ctx -> h[32,512]
    GEMM(ctx, 320, Pf(22), Pf(23), Pf(100), h, 512, 512, 320, 0);
    // 6) transformer layers: base = 24 + 12*l
    //    +0 W1.W +1 W1.b +2 W2.W +3 W2.b +4 Wo.W +5 Wo.b +6 Wv.W +7 Wv.b
    //    +8 ln1b +9 ln1g +10 ln2b +11 ln2g
    for (int l = 0; l < 6; ++l) {
        int p = 24 + 12 * l;
        GEMM(h,  512, Pf(p + 6), Pf(p + 7), nullptr, t1, 512, 512, 512, 0);   // v
        GEMM(t1, 512, Pf(p + 4), Pf(p + 5), nullptr, t2, 512, 512, 512, 0);   // attn
        add_ln_kernel<<<BATCH, 256, 0, stream>>>(h, t2, Pf(p + 9), Pf(p + 8));
        GEMM(h,  512, Pf(p + 0), Pf(p + 1), nullptr, t0, 2048, 2048, 512, 1); // ff hidden
        GEMM(t0, 2048, Pf(p + 2), Pf(p + 3), nullptr, t1, 512, 512, 2048, 0); // ff out
        add_ln_kernel<<<BATCH, 256, 0, stream>>>(h, t1, Pf(p + 11), Pf(p + 10));
    }
    // 7) heads -> d_out (concatenated in return order)
    //    op_head 96..99, color_map 4..7, translate 133..136, scale 111..114,
    //    extract 16..19, stop 125..128, conf 8..11
    GEMM(h, 512, Pf(96),  Pf(97),  nullptr, t0, 256, 256, 512, 1);
    GEMM(t0, 256, Pf(98), Pf(99),  nullptr, out + 0,    41, 41, 256, 0);
    GEMM(h, 512, Pf(4),   Pf(5),   nullptr, t0, 128, 128, 512, 1);
    GEMM(t0, 128, Pf(6),  Pf(7),   nullptr, out + 1312, 100, 100, 128, 0);
    GEMM(h, 512, Pf(133), Pf(134), nullptr, t0, 64, 64, 512, 1);
    GEMM(t0, 64, Pf(135), Pf(136), nullptr, out + 4512, 8, 8, 64, 0);
    GEMM(h, 512, Pf(111), Pf(112), nullptr, t0, 64, 64, 512, 1);
    GEMM(t0, 64, Pf(113), Pf(114), nullptr, out + 4768, 9, 9, 64, 0);
    GEMM(h, 512, Pf(16),  Pf(17),  nullptr, t0, 32, 32, 512, 1);
    GEMM(t0, 32, Pf(18),  Pf(19),  nullptr, out + 5056, 10, 10, 32, 0);
    GEMM(h, 512, Pf(125), Pf(126), nullptr, t0, 128, 128, 512, 1);
    GEMM(t0, 128, Pf(127), Pf(128), nullptr, out + 5376, 1, 1, 128, 2);
    GEMM(h, 512, Pf(8),   Pf(9),   nullptr, t0, 128, 128, 512, 1);
    GEMM(t0, 128, Pf(10), Pf(11),  nullptr, out + 5408, 1, 1, 128, 2);
}
